// VIM_MOS_47029891891690
// MI455X (gfx1250) — matrix operand order repaired and checked by reading, not compiled, not run on hardware
//
#include <hip/hip_runtime.h>
#include <hip/hip_bf16.h>
#include <math.h>

typedef __bf16 bf16;
typedef __attribute__((ext_vector_type(16))) __bf16 v16bf;
typedef __attribute__((ext_vector_type(8)))  __bf16 v8bf;
typedef __attribute__((ext_vector_type(8)))  float  v8f;

#define DEV __device__ __forceinline__

// Problem constants (from reference)
constexpr int Bb  = 4;
constexpr int Ll  = 512;
constexpr int DM  = 768;        // D_MODEL
constexpr int DI  = 1536;       // D_INNER
constexpr int NE  = 4;          // N_EXP
constexpr int DTR = 48;         // DT_RANK
constexpr int DTP = 64;         // DT_RANK padded to K%32==0
constexpr int XD  = 80;         // DT_RANK + 2*D_STATE
constexpr int MT  = Bb * Ll;    // 2048 tokens

DEV float sigmoidf_(float x) { return 1.f / (1.f + __expf(-x)); }
DEV float siluf_(float x)    { return x * sigmoidf_(x); }
DEV float softplusf_(float x){ return x > 20.f ? x : log1pf(expf(x)); }

// ---------------------------------------------------------------------------
// bf16 WMMA GEMM, 16x16 tile per wave (used for N not divisible by 64).
// C[M,N] = epilogue( A[M,K] * W[N,K]^T + bias )
// EPI: 0 = f32 store, 1 = bias + softplus -> f32, 2 = bias + SiLU -> bf16
// ---------------------------------------------------------------------------
template <int EPI>
__global__ __launch_bounds__(256) void gemm_bf16_wmma(
    const bf16* __restrict__ A, const bf16* __restrict__ W,
    const float* __restrict__ bias, float* __restrict__ Cf, bf16* __restrict__ Cb,
    int M, int N, int K)
{
    const int gtid  = blockIdx.x * blockDim.x + threadIdx.x;
    const int wave  = gtid >> 5;
    const int lane  = gtid & 31;
    const int ntile = N >> 4;
    const int mt    = wave / ntile;
    const int nt    = wave - mt * ntile;
    if (mt >= (M >> 4)) return;

    const int half = lane >> 4;   // lane group selects K sub-range per ISA layout
    const int r16  = lane & 15;

    const bf16* Ap = A + (size_t)(mt * 16 + r16) * K + half * 8;
    const bf16* Wp = W + (size_t)(nt * 16 + r16) * K + half * 8;

    v8f acc = {};
    for (int k0 = 0; k0 < K; k0 += 32) {
        union { v16bf v; v8bf h[2]; } a, b;
        a.h[0] = *(const v8bf*)(Ap + k0);        // K = base+0..7
        a.h[1] = *(const v8bf*)(Ap + k0 + 16);   // K = base+16..23
        b.h[0] = *(const v8bf*)(Wp + k0);        // K = kbase+0..7
        b.h[1] = *(const v8bf*)(Wp + k0 + 16);   // K = kbase+16..23
        __builtin_prefetch(Ap + k0 + 32, 0, 3);  // -> global_prefetch_b8
        __builtin_prefetch(Wp + k0 + 32, 0, 3);
        acc = __builtin_amdgcn_wmma_f32_16x16x32_bf16(
                  false, a.v, false, b.v, (short)0, acc, false, false);
    }

    const int col  = nt * 16 + r16;
    const int mrow = mt * 16 + half * 8;
#pragma unroll
    for (int r = 0; r < 8; ++r) {
        float v = acc[r];
        size_t idx = (size_t)(mrow + r) * N + col;
        if constexpr (EPI == 0) {
            Cf[idx] = v;
        } else if constexpr (EPI == 1) {
            Cf[idx] = softplusf_(v + bias[col]);
        } else {
            Cb[idx] = (bf16)siluf_(v + bias[col]);
        }
    }
}

// ---------------------------------------------------------------------------
// bf16 WMMA GEMM, 32x64 tile per wave (2 A-frags x 4 B-frags, 8 accumulators).
// Each A fragment is reused 4x and each B fragment 2x -> ~768 B of operand
// traffic per WMMA instead of ~2 KB, and 8 back-to-back v_wmma per K-step.
// Requires M % 32 == 0, N % 64 == 0, K % 32 == 0.
// ---------------------------------------------------------------------------
template <int EPI>
__global__ __launch_bounds__(256) void gemm_bf16_wmma_32x64(
    const bf16* __restrict__ A, const bf16* __restrict__ W,
    const float* __restrict__ bias, float* __restrict__ Cf, bf16* __restrict__ Cb,
    int M, int N, int K)
{
    const int gtid = blockIdx.x * blockDim.x + threadIdx.x;
    const int wave = gtid >> 5;
    const int lane = gtid & 31;
    const int nt64 = N >> 6;
    const int mt   = wave / nt64;
    const int nt   = wave - mt * nt64;
    if (mt >= (M >> 5)) return;

    const int half = lane >> 4;
    const int r16  = lane & 15;

    const bf16* Ap[2];
    Ap[0] = A + (size_t)(mt * 32 + r16) * K + half * 8;
    Ap[1] = Ap[0] + (size_t)16 * K;
    const bf16* Wp[4];
    Wp[0] = W + (size_t)(nt * 64 + r16) * K + half * 8;
    Wp[1] = Wp[0] + (size_t)16 * K;
    Wp[2] = Wp[0] + (size_t)32 * K;
    Wp[3] = Wp[0] + (size_t)48 * K;

    v8f acc[2][4] = {};
    for (int k0 = 0; k0 < K; k0 += 32) {
        union { v16bf v; v8bf h[2]; } a[2], b[4];
#pragma unroll
        for (int i = 0; i < 2; ++i) {
            a[i].h[0] = *(const v8bf*)(Ap[i] + k0);
            a[i].h[1] = *(const v8bf*)(Ap[i] + k0 + 16);
            __builtin_prefetch(Ap[i] + k0 + 32, 0, 3);
        }
#pragma unroll
        for (int j = 0; j < 4; ++j) {
            b[j].h[0] = *(const v8bf*)(Wp[j] + k0);
            b[j].h[1] = *(const v8bf*)(Wp[j] + k0 + 16);
            __builtin_prefetch(Wp[j] + k0 + 32, 0, 3);
        }
#pragma unroll
        for (int i = 0; i < 2; ++i)
#pragma unroll
            for (int j = 0; j < 4; ++j)
                acc[i][j] = __builtin_amdgcn_wmma_f32_16x16x32_bf16(
                    false, a[i].v, false, b[j].v, (short)0, acc[i][j], false, false);
    }

#pragma unroll
    for (int i = 0; i < 2; ++i) {
        const int mrow = mt * 32 + i * 16 + half * 8;
#pragma unroll
        for (int j = 0; j < 4; ++j) {
            const int col = nt * 64 + j * 16 + r16;
#pragma unroll
            for (int r = 0; r < 8; ++r) {
                float v = acc[i][j][r];
                size_t idx = (size_t)(mrow + r) * N + col;
                if constexpr (EPI == 0) {
                    Cf[idx] = v;
                } else if constexpr (EPI == 1) {
                    Cf[idx] = softplusf_(v + bias[col]);
                } else {
                    Cb[idx] = (bf16)siluf_(v + bias[col]);
                }
            }
        }
    }
}

// ---------------------------------------------------------------------------
// Elementwise converts
// ---------------------------------------------------------------------------
__global__ __launch_bounds__(256) void f32_to_bf16_k(
    const float* __restrict__ s, bf16* __restrict__ d, int n)
{
    int i = blockIdx.x * blockDim.x + threadIdx.x;
    if (i < n) d[i] = (bf16)s[i];
}

// dst[r, c<take] = (bf16)src[r*scols + c]; else 0   (K-padding for WMMA)
__global__ __launch_bounds__(256) void pad_convert_bf16_k(
    const float* __restrict__ s, bf16* __restrict__ d,
    int rows, int scols, int dcols, int take)
{
    int i = blockIdx.x * blockDim.x + threadIdx.x;
    if (i >= rows * dcols) return;
    int r = i / dcols, c = i - r * dcols;
    d[i] = (c < take) ? (bf16)s[(size_t)r * scols + c] : (bf16)0.f;
}

__global__ __launch_bounds__(256) void neg_exp_k(
    const float* __restrict__ s, float* __restrict__ d, int n)
{
    int i = blockIdx.x * blockDim.x + threadIdx.x;
    if (i < n) d[i] = -__expf(s[i]);
}

// ---------------------------------------------------------------------------
// Fused fwd+bwd depthwise causal conv (width 4) + SiLU.
// xz layout: (B, L, 2*DI) tokens-major. Backward branch uses the same first-DI
// channels at time-reversed positions.
// ---------------------------------------------------------------------------
__global__ __launch_bounds__(256) void conv_silu_k(
    const float* __restrict__ xz,
    const float* __restrict__ cw,  const float* __restrict__ cb,
    const float* __restrict__ cwb, const float* __restrict__ cbb,
    float* __restrict__ xc,  bf16* __restrict__ xcbf,
    float* __restrict__ xr,  bf16* __restrict__ xrbf)
{
    int i = blockIdx.x * blockDim.x + threadIdx.x;
    if (i >= Bb * Ll * DI) return;
    int d = i % DI;
    int l = (i / DI) % Ll;
    int b = i / (DI * Ll);

    float accf = cb[d], accb = cbb[d];
#pragma unroll
    for (int j = 0; j < 4; ++j) {
        int li = l - 3 + j;
        if (li >= 0) {
            accf += xz[((size_t)(b * Ll + li)) * (2 * DI) + d] * cw[d * 4 + j];
            accb += xz[((size_t)(b * Ll + (Ll - 1 - li))) * (2 * DI) + d] * cwb[d * 4 + j];
        }
    }
    float yf = siluf_(accf), yb = siluf_(accb);
    size_t o = (size_t)(b * Ll + l) * DI + d;
    xc[o] = yf; xcbf[o] = (bf16)yf;
    xr[o] = yb; xrbf[o] = (bf16)yb;
}

// ---------------------------------------------------------------------------
// Expert gate: one wave32 per token. gate = sigmoid(h1 @ w2^T + b2); softmax.
// ---------------------------------------------------------------------------
__global__ __launch_bounds__(256) void gate_softmax_k(
    const bf16* __restrict__ h1, const float* __restrict__ w2,
    const float* __restrict__ b2, float* __restrict__ wout)
{
    int gtid = blockIdx.x * blockDim.x + threadIdx.x;
    int t = gtid >> 5, lane = gtid & 31;
    if (t >= MT) return;
    const bf16* hp = h1 + (size_t)t * DI;
    float a0 = 0, a1 = 0, a2 = 0, a3 = 0;
    for (int k = lane; k < DI; k += 32) {
        float hv = (float)hp[k];
        a0 += hv * w2[k];
        a1 += hv * w2[DI + k];
        a2 += hv * w2[2 * DI + k];
        a3 += hv * w2[3 * DI + k];
    }
#pragma unroll
    for (int off = 16; off > 0; off >>= 1) {
        a0 += __shfl_xor(a0, off, 32);
        a1 += __shfl_xor(a1, off, 32);
        a2 += __shfl_xor(a2, off, 32);
        a3 += __shfl_xor(a3, off, 32);
    }
    if (lane == 0) {
        float g0 = sigmoidf_(a0 + b2[0]), g1 = sigmoidf_(a1 + b2[1]);
        float g2 = sigmoidf_(a2 + b2[2]), g3 = sigmoidf_(a3 + b2[3]);
        float m  = fmaxf(fmaxf(g0, g1), fmaxf(g2, g3));
        float e0 = __expf(g0 - m), e1 = __expf(g1 - m);
        float e2 = __expf(g2 - m), e3 = __expf(g3 - m);
        float inv = 1.f / (e0 + e1 + e2 + e3);
        float* wp = wout + (size_t)t * 4;
        wp[0] = e0 * inv; wp[1] = e1 * inv; wp[2] = e2 * inv; wp[3] = e3 * inv;
    }
}

// ---------------------------------------------------------------------------
// Selective scan: thread = (b, d, e) owns 4 states; h = exp(delta*A)*h + delta*B*x.
// Expert contribution weighted by gate and quad-reduced with wave32 shuffles,
// eliminating the (B,L,DI,NE) intermediate entirely.
// ---------------------------------------------------------------------------
__global__ __launch_bounds__(256) void selective_scan_k(
    const float* __restrict__ delta, const float* __restrict__ xdbl,
    const float* __restrict__ xc,    const float* __restrict__ An,
    const float* __restrict__ wgt,   float* __restrict__ yout)
{
    int tid = blockIdx.x * blockDim.x + threadIdx.x;
    if (tid >= Bb * DI * NE) return;
    int e = tid & 3;
    int d = (tid >> 2) % DI;
    int b = tid / (DI * NE);

    const float* Ap = An + d * 16 + e * 4;
    float A0 = Ap[0], A1 = Ap[1], A2 = Ap[2], A3 = Ap[3];
    float h0 = 0, h1 = 0, h2 = 0, h3 = 0;

    for (int l = 0; l < Ll; ++l) {
        size_t t  = (size_t)b * Ll + l;
        float dl  = delta[t * DI + d];
        float xv  = xc[t * DI + d];
        const float* Bp = xdbl + t * XD + DTR + e * 4;       // B block
        const float* Cp = xdbl + t * XD + DTR + 16 + e * 4;  // C block
        float dx = dl * xv;
        h0 = __expf(dl * A0) * h0 + dx * Bp[0];
        h1 = __expf(dl * A1) * h1 + dx * Bp[1];
        h2 = __expf(dl * A2) * h2 + dx * Bp[2];
        h3 = __expf(dl * A3) * h3 + dx * Bp[3];
        float p = (h0 * Cp[0] + h1 * Cp[1] + h2 * Cp[2] + h3 * Cp[3]) * wgt[t * 4 + e];
        p += __shfl_xor(p, 1, 32);
        p += __shfl_xor(p, 2, 32);
        if (e == 0) yout[t * DI + d] = p;
    }
}

// ---------------------------------------------------------------------------
// Combine fwd + time-reversed bwd; silu(zb[L-1-l']) == silu(z[l]) so one gate.
// ---------------------------------------------------------------------------
__global__ __launch_bounds__(256) void combine_k(
    const float* __restrict__ yf,  const float* __restrict__ yb,
    const float* __restrict__ xc,  const float* __restrict__ xcb,
    const float* __restrict__ xz,  const float* __restrict__ Dv,
    const float* __restrict__ Dbv, bf16* __restrict__ ysum)
{
    int i = blockIdx.x * blockDim.x + threadIdx.x;
    if (i >= MT * DI) return;
    int d = i % DI;
    int l = (i / DI) % Ll;
    int b = i / (DI * Ll);
    size_t t  = (size_t)b * Ll + l;
    size_t tr = (size_t)b * Ll + (Ll - 1 - l);
    float y  = yf[t * DI + d]  + xc[t * DI + d]  * Dv[d];
    float y2 = yb[tr * DI + d] + xcb[tr * DI + d] * Dbv[d];
    float z  = xz[t * (2 * DI) + DI + d];
    ysum[t * DI + d] = (bf16)((y + y2) * siluf_(z));
}

// ---------------------------------------------------------------------------
extern "C" void kernel_launch(void* const* d_in, const int* in_sizes, int n_in,
                              void* d_out, int out_size, void* d_ws, size_t ws_size,
                              hipStream_t stream)
{
    (void)in_sizes; (void)n_in; (void)out_size; (void)ws_size;
    const float* hid   = (const float*)d_in[0];
    const float* winp  = (const float*)d_in[1];
    const float* cw    = (const float*)d_in[2];
    const float* cb    = (const float*)d_in[3];
    const float* cwb   = (const float*)d_in[4];
    const float* cbb   = (const float*)d_in[5];
    const float* wxp   = (const float*)d_in[6];
    const float* wxpb  = (const float*)d_in[7];
    const float* wdt   = (const float*)d_in[8];
    const float* bdt   = (const float*)d_in[9];
    const float* wdtb  = (const float*)d_in[10];
    const float* bdtb  = (const float*)d_in[11];
    const float* Alog  = (const float*)d_in[12];
    const float* Ablog = (const float*)d_in[13];
    const float* Dv    = (const float*)d_in[14];
    const float* Dbv   = (const float*)d_in[15];
    const float* w1    = (const float*)d_in[16];
    const float* b1    = (const float*)d_in[17];
    const float* w2    = (const float*)d_in[18];
    const float* b2    = (const float*)d_in[19];
    const float* wo    = (const float*)d_in[20];
    float* out = (float*)d_out;

    // Workspace carve-out (256B aligned)
    char* p = (char*)d_ws;
    auto alloc = [&](size_t bytes) -> char* {
        char* r = p; p += (bytes + 255) & ~(size_t)255; return r;
    };
    bf16* hid_bf   = (bf16*)alloc((size_t)MT * DM * 2);
    bf16* winp_bf  = (bf16*)alloc((size_t)2 * DI * DM * 2);
    bf16* wxp_bf   = (bf16*)alloc((size_t)XD * DI * 2);
    bf16* wxpb_bf  = (bf16*)alloc((size_t)XD * DI * 2);
    bf16* wdt_bf   = (bf16*)alloc((size_t)DI * DTP * 2);
    bf16* wdtb_bf  = (bf16*)alloc((size_t)DI * DTP * 2);
    bf16* w1_bf    = (bf16*)alloc((size_t)DI * DI * 2);
    bf16* wo_bf    = (bf16*)alloc((size_t)DM * DI * 2);
    float* xz      = (float*)alloc((size_t)MT * 2 * DI * 4);
    float* xc      = (float*)alloc((size_t)MT * DI * 4);
    bf16*  xc_bf   = (bf16*)alloc((size_t)MT * DI * 2);
    float* xcb     = (float*)alloc((size_t)MT * DI * 4);
    bf16*  xcb_bf  = (bf16*)alloc((size_t)MT * DI * 2);
    float* xdbl0   = (float*)alloc((size_t)MT * XD * 4);
    float* xdbl1   = (float*)alloc((size_t)MT * XD * 4);
    bf16*  dtr0_bf = (bf16*)alloc((size_t)MT * DTP * 2);
    bf16*  dtr1_bf = (bf16*)alloc((size_t)MT * DTP * 2);
    float* delta0  = (float*)alloc((size_t)MT * DI * 4);
    float* delta1  = (float*)alloc((size_t)MT * DI * 4);
    float* An0     = (float*)alloc((size_t)DI * 16 * 4);
    float* An1     = (float*)alloc((size_t)DI * 16 * 4);
    bf16*  h1_bf   = (bf16*)alloc((size_t)MT * DI * 2);
    float* wgt     = (float*)alloc((size_t)MT * 4 * 4);
    float* yscan0  = (float*)alloc((size_t)MT * DI * 4);
    float* yscan1  = (float*)alloc((size_t)MT * DI * 4);
    bf16*  ysum_bf = (bf16*)alloc((size_t)MT * DI * 2);

    auto nb = [](size_t n) { return (unsigned)((n + 255) / 256); };
    auto grid16 = [](int M, int N) {
        return (unsigned)(((size_t)(M / 16) * (N / 16) * 32 + 255) / 256);
    };
    auto grid32x64 = [](int M, int N) {
        return (unsigned)(((size_t)(M / 32) * (N / 64) * 32 + 255) / 256);
    };

    // 1) Conversions
    f32_to_bf16_k<<<nb((size_t)MT * DM), 256, 0, stream>>>(hid, hid_bf, MT * DM);
    f32_to_bf16_k<<<nb((size_t)2 * DI * DM), 256, 0, stream>>>(winp, winp_bf, 2 * DI * DM);
    f32_to_bf16_k<<<nb((size_t)XD * DI), 256, 0, stream>>>(wxp, wxp_bf, XD * DI);
    f32_to_bf16_k<<<nb((size_t)XD * DI), 256, 0, stream>>>(wxpb, wxpb_bf, XD * DI);
    f32_to_bf16_k<<<nb((size_t)DI * DI), 256, 0, stream>>>(w1, w1_bf, DI * DI);
    f32_to_bf16_k<<<nb((size_t)DM * DI), 256, 0, stream>>>(wo, wo_bf, DM * DI);
    pad_convert_bf16_k<<<nb((size_t)DI * DTP), 256, 0, stream>>>(wdt,  wdt_bf,  DI, DTR, DTP, DTR);
    pad_convert_bf16_k<<<nb((size_t)DI * DTP), 256, 0, stream>>>(wdtb, wdtb_bf, DI, DTR, DTP, DTR);
    neg_exp_k<<<nb((size_t)DI * 16), 256, 0, stream>>>(Alog,  An0, DI * 16);
    neg_exp_k<<<nb((size_t)DI * 16), 256, 0, stream>>>(Ablog, An1, DI * 16);

    // 2) in_proj: xz[t, 0..3071]   (32x64 tiles)
    gemm_bf16_wmma_32x64<0><<<grid32x64(MT, 2 * DI), 256, 0, stream>>>(
        hid_bf, winp_bf, nullptr, xz, nullptr, MT, 2 * DI, DM);

    // 3) Fused fwd+bwd causal conv + SiLU
    conv_silu_k<<<nb((size_t)MT * DI), 256, 0, stream>>>(
        xz, cw, cb, cwb, cbb, xc, xc_bf, xcb, xcb_bf);

    // 4) MLP branch: h1 = silu(xc @ w1^T + b1)  (bf16 out), then gate softmax
    gemm_bf16_wmma_32x64<2><<<grid32x64(MT, DI), 256, 0, stream>>>(
        xc_bf, w1_bf, b1, nullptr, h1_bf, MT, DI, DI);
    gate_softmax_k<<<nb((size_t)MT * 32), 256, 0, stream>>>(h1_bf, w2, b2, wgt);

    // 5) x_proj (fwd + bwd): x_dbl = xc @ Wx^T  (N=80 -> 16x16 tiles)
    gemm_bf16_wmma<0><<<grid16(MT, XD), 256, 0, stream>>>(
        xc_bf, wxp_bf, nullptr, xdbl0, nullptr, MT, XD, DI);
    gemm_bf16_wmma<0><<<grid16(MT, XD), 256, 0, stream>>>(
        xcb_bf, wxpb_bf, nullptr, xdbl1, nullptr, MT, XD, DI);

    // 6) dtr (first 48 cols of x_dbl) -> padded bf16; delta = softplus(dtr @ Wdt^T + bdt)
    pad_convert_bf16_k<<<nb((size_t)MT * DTP), 256, 0, stream>>>(xdbl0, dtr0_bf, MT, XD, DTP, DTR);
    pad_convert_bf16_k<<<nb((size_t)MT * DTP), 256, 0, stream>>>(xdbl1, dtr1_bf, MT, XD, DTP, DTR);
    gemm_bf16_wmma_32x64<1><<<grid32x64(MT, DI), 256, 0, stream>>>(
        dtr0_bf, wdt_bf, bdt, delta0, nullptr, MT, DI, DTP);
    gemm_bf16_wmma_32x64<1><<<grid32x64(MT, DI), 256, 0, stream>>>(
        dtr1_bf, wdtb_bf, bdtb, delta1, nullptr, MT, DI, DTP);

    // 7) Selective scans (gate-weighted, quad-reduced)
    selective_scan_k<<<nb((size_t)Bb * DI * NE), 256, 0, stream>>>(
        delta0, xdbl0, xc, An0, wgt, yscan0);
    selective_scan_k<<<nb((size_t)Bb * DI * NE), 256, 0, stream>>>(
        delta1, xdbl1, xcb, An1, wgt, yscan1);

    // 8) Combine fwd + reversed bwd, gate with silu(z), emit bf16
    combine_k<<<nb((size_t)MT * DI), 256, 0, stream>>>(
        yscan0, yscan1, xc, xcb, xz, Dv, Dbv, ysum_bf);

    // 9) out_proj -> d_out (f32)   (32x64 tiles)
    gemm_bf16_wmma_32x64<0><<<grid32x64(MT, DM), 256, 0, stream>>>(
        ysum_bf, wo_bf, nullptr, out, nullptr, MT, DM, DI);
}
